// GraphNet_19026705121528
// MI455X (gfx1250) — matrix operand order repaired and checked by reading, not compiled, not run on hardware
//
#include <hip/hip_runtime.h>

// ---------------------------------------------------------------------------
// GraphNet on gfx1250: f16 WMMA tile pipeline (memory-bound workload).
// ---------------------------------------------------------------------------

typedef __attribute__((ext_vector_type(16))) _Float16 v16h;
typedef __attribute__((ext_vector_type(8)))  _Float16 v8h;
typedef __attribute__((ext_vector_type(8)))  float    v8f;
typedef __attribute__((ext_vector_type(4)))  float    v4f;

#define WPB 8  // waves per 256-thread block (wave32)

__device__ __forceinline__ v8f wmma_f16(v16h a, v16h b, v8f c) {
    // D = A(16x32 f16) * B(32x16 f16) + C(16x16 f32)
    return __builtin_amdgcn_wmma_f32_16x16x32_f16(
        /*neg_a=*/false, a, /*neg_b=*/false, b,
        /*c_mod=*/(short)0, c, /*reuse_a=*/false, /*reuse_b=*/false);
}

// Load B operand: weight matrix W[KROWS][5] (f32 row-major), zero-padded to
// 32x16 f16. Layout: lanes 0-15 hold K in {0..7,16..23}; lanes 16-31 hold
// K in {8..15,24..31} (elem j -> K=kb+j, j>=8 -> K=kb+8+j); column = lane&15.
template <int KROWS>
__device__ __forceinline__ v16h load_B(const float* __restrict__ W, int lane) {
    const int n  = lane & 15;
    const int kb = (lane >> 4) << 3;  // 0 or 8
    v16h b;
#pragma unroll
    for (int j = 0; j < 16; ++j) {
        const int k = kb + j + (j >= 8 ? 8 : 0);
        float w = 0.0f;
        if (n < 5 && k < KROWS) w = W[k * 5 + n];
        b[j] = (_Float16)w;
    }
    return b;
}

// Load A operand from f32 global row-major [rows][32], one 16x32 tile.
// Layout: lane&15 = row M; lanes 0-15: K in {0..7,16..23}; lanes 16-31:
// K in {8..15,24..31}. Elements 0..7 -> K=kb+j, 8..15 -> K=kb+16+(j-8).
__device__ __forceinline__ v16h load_A_rows(const float* __restrict__ feats,
                                            long row, int lane) {
    const int kb = (lane >> 4) ? 8 : 0;
    const float* p = feats + row * 32 + kb;
    v4f c0 = __builtin_nontemporal_load((const v4f*)(p));
    v4f c1 = __builtin_nontemporal_load((const v4f*)(p + 4));
    v4f c2 = __builtin_nontemporal_load((const v4f*)(p + 16));
    v4f c3 = __builtin_nontemporal_load((const v4f*)(p + 20));
    v16h a;
#pragma unroll
    for (int j = 0; j < 4; ++j) {
        a[j]      = (_Float16)c0[j];
        a[4 + j]  = (_Float16)c1[j];
        a[8 + j]  = (_Float16)c2[j];
        a[12 + j] = (_Float16)c3[j];
    }
    return a;
}

// Load A operand from a per-wave 16x32 f16 LDS tile (row stride 32 halfs).
__device__ __forceinline__ v16h load_A_lds(const _Float16* lds, int lane) {
    const int r  = lane & 15;
    const int kb = (lane >> 4) ? 8 : 0;
    const _Float16* p = lds + r * 32 + kb;
    v8h lo = *(const v8h*)(p);       // ds_load_b128
    v8h hi = *(const v8h*)(p + 16);  // ds_load_b128
    v16h a;
#pragma unroll
    for (int j = 0; j < 8; ++j) { a[j] = lo[j]; a[8 + j] = hi[j]; }
    return a;
}

__device__ __forceinline__ void lds_fence() {
    __builtin_amdgcn_wave_barrier();
    asm volatile("s_wait_dscnt 0x0" ::: "memory");  // LDS in-order per wave; block compiler reorder
}

// ---------------------------------------------------------------------------
// g = global_feats(1x32) @ W_g(32x5) + b_g   (one wave)
// ---------------------------------------------------------------------------
__global__ void glob_embed_kernel(const float* __restrict__ gf,
                                  const float* __restrict__ W,
                                  const float* __restrict__ b,
                                  float* __restrict__ g) {
    const int lane = threadIdx.x;
    const float x = gf[lane];
#pragma unroll
    for (int c = 0; c < 5; ++c) {
        float p = x * W[lane * 5 + c];
#pragma unroll
        for (int off = 16; off > 0; off >>= 1) p += __shfl_xor(p, off, 32);
        if (lane == 0) g[c] = p + b[c];
    }
}

// ---------------------------------------------------------------------------
// n_emb[N][8] = node_feats(Nx32) @ W_n(32x5) + b_n   (WMMA, 16 nodes/tile)
// ---------------------------------------------------------------------------
__global__ __launch_bounds__(256) void node_embed_kernel(
    const float* __restrict__ node_feats, const float* __restrict__ W,
    const float* __restrict__ b, float* __restrict__ n_emb, int N) {
    const int lane = threadIdx.x & 31;
    const int wave = blockIdx.x * WPB + (threadIdx.x >> 5);
    const int nW   = gridDim.x * WPB;
    const int n    = lane & 15;
    const int grp  = lane >> 4;

    const v16h B  = load_B<32>(W, lane);
    const float bn_ = (n < 5) ? b[n] : 0.0f;

    const int nTiles = (N + 15) >> 4;
    for (int t = wave; t < nTiles; t += nW) {
        const int base = t << 4;
        long row = base + n; if (row >= N) row = N - 1;
        v16h a = load_A_rows(node_feats, row, lane);
        v8f d = {};
        d = wmma_f16(a, B, d);
        if (n < 5) {
            float* p = n_emb + (long)(base + grp * 8) * 8 + n;
            if (base + 16 <= N) {           // full tile: unguarded, imm offsets
#pragma unroll
                for (int v = 0; v < 8; ++v) p[v * 8] = d[v] + bn_;
            } else {
#pragma unroll
                for (int v = 0; v < 8; ++v)
                    if (base + grp * 8 + v < N) p[v * 8] = d[v] + bn_;
            }
        }
    }
}

// ---------------------------------------------------------------------------
// Edge pipeline: embed -> gather -> MLP(20->5->5) -> atomic scatter
// ---------------------------------------------------------------------------
__global__ __launch_bounds__(256) void edge_kernel(
    const float* __restrict__ edge_feats, const int* __restrict__ senders,
    const int* __restrict__ receivers,
    const float* __restrict__ W_e, const float* __restrict__ b_e,
    const float* __restrict__ We1, const float* __restrict__ be1,
    const float* __restrict__ We2, const float* __restrict__ be2,
    const float* __restrict__ n_emb, const float* __restrict__ g_emb,
    float* __restrict__ sent_agg, float* __restrict__ recv_agg, int E) {
    __shared__ _Float16 tiles[WPB][16 * 32];
    const int lane = threadIdx.x & 31;
    const int wIn  = threadIdx.x >> 5;
    _Float16* lds  = tiles[wIn];
    const int wave = blockIdx.x * WPB + wIn;
    const int nW   = gridDim.x * WPB;
    const int n    = lane & 15;
    const int grp  = lane >> 4;

    const v16h Bemb = load_B<32>(W_e, lane);
    const v16h B1   = load_B<20>(We1, lane);
    const v16h B2   = load_B<5>(We2, lane);
    const float bE = (n < 5) ? b_e[n] : 0.0f;
    const float b1 = (n < 5) ? be1[n] : 0.0f;
    const float b2 = (n < 5) ? be2[n] : 0.0f;
    float gv[5];
#pragma unroll
    for (int c = 0; c < 5; ++c) gv[c] = g_emb[c];

    for (int i = lane; i < 16 * 32; i += 32) lds[i] = (_Float16)0.0f;
    lds_fence();

    const int nTiles = (E + 15) >> 4;
    for (int t = wave; t < nTiles; t += nW) {
        const int base = t << 4;
        // -- 1: edge embedding (K=32 WMMA), write cols [0,5) of LDS tile --
        long row = base + n; if (row >= E) row = E - 1;
        v16h a = load_A_rows(edge_feats, row, lane);
        v8f e = {};
        e = wmma_f16(a, Bemb, e);
        if (n < 5) {
            _Float16* lp = lds + grp * 8 * 32 + n;
#pragma unroll
            for (int v = 0; v < 8; ++v)
                lp[v * 32] = (_Float16)(e[v] + bE);
        }
        // -- 2: gather sender (grp0 -> cols [5,10)) / receiver (grp1 -> [10,15)) --
        {
            int eIdx = base + n; if (eIdx >= E) eIdx = E - 1;
            const int node = grp ? receivers[eIdx] : senders[eIdx];
            const float* src = n_emb + (long)node * 8;
            v4f q = *(const v4f*)src;
            const float q4 = src[4];
            _Float16* lp = lds + n * 32 + (grp ? 10 : 5);
            lp[0] = (_Float16)q[0];
            lp[1] = (_Float16)q[1];
            lp[2] = (_Float16)q[2];
            lp[3] = (_Float16)q[3];
            lp[4] = (_Float16)q4;
        }
        // -- 3: broadcast globals into cols [15,20) --
        if (!grp) {
            _Float16* lp = lds + n * 32 + 15;
#pragma unroll
            for (int c = 0; c < 5; ++c) lp[c] = (_Float16)gv[c];
        }
        lds_fence();
        // -- 4: MLP layer 1 (K=20 zero-padded WMMA) --
        v16h a1 = load_A_lds(lds, lane);
        v8f h = {};
        h = wmma_f16(a1, B1, h);
        lds_fence();
        // relu + bias back into cols [0,5), zero cols [5,20)
        {
            _Float16* lp = lds + grp * 8 * 32;
#pragma unroll
            for (int v = 0; v < 8; ++v) {
                _Float16 hv = (_Float16)0.0f;
                if (n < 5) hv = (_Float16)fmaxf(h[v] + b1, 0.0f);
                lp[v * 32 + n] = hv;
                if (n < 4) lp[v * 32 + 16 + n] = (_Float16)0.0f;
            }
        }
        lds_fence();
        // -- 5: MLP layer 2 (K=5 zero-padded WMMA) --
        v16h a2 = load_A_lds(lds, lane);
        v8f o = {};
        o = wmma_f16(a2, B2, o);
        // -- 6: segment-sum scatter: fire-and-forget f32 atomics --
        if (n < 5) {
            const int mBase = base + grp * 8;
            const int* sp = senders + mBase;
            const int* rp = receivers + mBase;
            if (base + 16 <= E) {           // full tile: no per-edge guards
#pragma unroll
                for (int v = 0; v < 8; ++v) {
                    const float val = fmaxf(o[v] + b2, 0.0f);
                    atomicAdd(&sent_agg[(long)sp[v] * 8 + n], val);
                    atomicAdd(&recv_agg[(long)rp[v] * 8 + n], val);
                }
            } else {
#pragma unroll
                for (int v = 0; v < 8; ++v) {
                    if (mBase + v < E) {
                        const float val = fmaxf(o[v] + b2, 0.0f);
                        atomicAdd(&sent_agg[(long)sp[v] * 8 + n], val);
                        atomicAdd(&recv_agg[(long)rp[v] * 8 + n], val);
                    }
                }
            }
        }
        lds_fence();
    }
}

// ---------------------------------------------------------------------------
// Node update: concat(n, sent, recv, g) -> MLP(20->5->5) -> out[N][5]
// ---------------------------------------------------------------------------
__global__ __launch_bounds__(256) void node_update_kernel(
    const float* __restrict__ n_emb, const float* __restrict__ sent_agg,
    const float* __restrict__ recv_agg, const float* __restrict__ g_emb,
    const float* __restrict__ Wn1, const float* __restrict__ bn1,
    const float* __restrict__ Wn2, const float* __restrict__ bn2,
    float* __restrict__ out, int N) {
    __shared__ _Float16 tiles[WPB][16 * 32];
    const int lane = threadIdx.x & 31;
    const int wIn  = threadIdx.x >> 5;
    _Float16* lds  = tiles[wIn];
    const int wave = blockIdx.x * WPB + wIn;
    const int nW   = gridDim.x * WPB;
    const int n    = lane & 15;
    const int grp  = lane >> 4;

    const v16h B1 = load_B<20>(Wn1, lane);
    const v16h B2 = load_B<5>(Wn2, lane);
    const float b1 = (n < 5) ? bn1[n] : 0.0f;
    const float b2 = (n < 5) ? bn2[n] : 0.0f;
    float gv[5];
#pragma unroll
    for (int c = 0; c < 5; ++c) gv[c] = g_emb[c];

    for (int i = lane; i < 16 * 32; i += 32) lds[i] = (_Float16)0.0f;
    lds_fence();

    const int nTiles = (N + 15) >> 4;
    for (int t = wave; t < nTiles; t += nW) {
        const int base = t << 4;
        int rowIdx = base + n; if (rowIdx >= N) rowIdx = N - 1;
        // grp0: n_emb -> cols [0,5);   grp1: sent -> cols [5,10)
        {
            const float* src = grp ? (sent_agg + (long)rowIdx * 8)
                                   : (n_emb + (long)rowIdx * 8);
            v4f q = *(const v4f*)src;
            const float q4 = src[4];
            _Float16* lp = lds + n * 32 + (grp ? 5 : 0);
            lp[0] = (_Float16)q[0];
            lp[1] = (_Float16)q[1];
            lp[2] = (_Float16)q[2];
            lp[3] = (_Float16)q[3];
            lp[4] = (_Float16)q4;
        }
        // grp0: recv -> cols [10,15);  grp1: globals -> cols [15,20)
        if (!grp) {
            const float* src = recv_agg + (long)rowIdx * 8;
            v4f q = *(const v4f*)src;
            const float q4 = src[4];
            _Float16* lp = lds + n * 32 + 10;
            lp[0] = (_Float16)q[0];
            lp[1] = (_Float16)q[1];
            lp[2] = (_Float16)q[2];
            lp[3] = (_Float16)q[3];
            lp[4] = (_Float16)q4;
        } else {
            _Float16* lp = lds + n * 32 + 15;
#pragma unroll
            for (int c = 0; c < 5; ++c) lp[c] = (_Float16)gv[c];
        }
        lds_fence();
        v16h a1 = load_A_lds(lds, lane);
        v8f h = {};
        h = wmma_f16(a1, B1, h);
        lds_fence();
        {
            _Float16* lp = lds + grp * 8 * 32;
#pragma unroll
            for (int v = 0; v < 8; ++v) {
                _Float16 hv = (_Float16)0.0f;
                if (n < 5) hv = (_Float16)fmaxf(h[v] + b1, 0.0f);
                lp[v * 32 + n] = hv;
                if (n < 4) lp[v * 32 + 16 + n] = (_Float16)0.0f;
            }
        }
        lds_fence();
        v16h a2 = load_A_lds(lds, lane);
        v8f o = {};
        o = wmma_f16(a2, B2, o);
        if (n < 5) {
            float* p = out + (long)(base + grp * 8) * 5 + n;
            if (base + 16 <= N) {           // full tile: unguarded, imm offsets
#pragma unroll
                for (int v = 0; v < 8; ++v) p[v * 5] = fmaxf(o[v] + b2, 0.0f);
            } else {
#pragma unroll
                for (int v = 0; v < 8; ++v)
                    if (base + grp * 8 + v < N) p[v * 5] = fmaxf(o[v] + b2, 0.0f);
            }
        }
        lds_fence();
    }
}

// ---------------------------------------------------------------------------
extern "C" void kernel_launch(void* const* d_in, const int* in_sizes, int n_in,
                              void* d_out, int out_size, void* d_ws, size_t ws_size,
                              hipStream_t stream) {
    const float* node_feats   = (const float*)d_in[0];
    const float* edge_feats   = (const float*)d_in[1];
    const float* global_feats = (const float*)d_in[2];
    const int*   senders      = (const int*)d_in[3];
    const int*   receivers    = (const int*)d_in[4];
    const float* W_e = (const float*)d_in[5];  const float* b_e = (const float*)d_in[6];
    const float* W_n = (const float*)d_in[7];  const float* b_n = (const float*)d_in[8];
    const float* W_g = (const float*)d_in[9];  const float* b_g = (const float*)d_in[10];
    const float* We1 = (const float*)d_in[11]; const float* be1 = (const float*)d_in[12];
    const float* We2 = (const float*)d_in[13]; const float* be2 = (const float*)d_in[14];
    const float* Wn1 = (const float*)d_in[15]; const float* bn1 = (const float*)d_in[16];
    const float* Wn2 = (const float*)d_in[17]; const float* bn2 = (const float*)d_in[18];

    const int N = in_sizes[0] / 32;
    const int E = in_sizes[3];

    // Workspace: n_emb[N*8] | sent[N*8] | recv[N*8] | g[8]  (stride 8 for alignment)
    float* ws    = (float*)d_ws;
    float* n_emb = ws;
    float* sent  = ws + (size_t)N * 8;
    float* recv  = ws + (size_t)N * 16;
    float* g     = ws + (size_t)N * 24;

    // Zero both aggregation buffers (contiguous) every call — deterministic.
    (void)hipMemsetAsync(sent, 0, (size_t)N * 16 * sizeof(float), stream);

    glob_embed_kernel<<<1, 32, 0, stream>>>(global_feats, W_g, b_g, g);

    const int nodeTiles  = (N + 15) / 16;
    int nodeBlocks = (nodeTiles + WPB - 1) / WPB;
    if (nodeBlocks > 1024) nodeBlocks = 1024;
    node_embed_kernel<<<nodeBlocks, 256, 0, stream>>>(node_feats, W_n, b_n, n_emb, N);

    const int edgeTiles = (E + 15) / 16;
    int edgeBlocks = (edgeTiles + WPB - 1) / WPB;
    if (edgeBlocks > 1280) edgeBlocks = 1280;
    edge_kernel<<<edgeBlocks, 256, 0, stream>>>(edge_feats, senders, receivers,
        W_e, b_e, We1, be1, We2, be2, n_emb, g, sent, recv, E);

    node_update_kernel<<<nodeBlocks, 256, 0, stream>>>(n_emb, sent, recv, g,
        Wn1, bn1, Wn2, bn2, (float*)d_out, N);
}
